// PerResidueLDDTGraphPred_1915555414559
// MI455X (gfx1250) — matrix operand order repaired and checked by reading, not compiled, not run on hardware
//
#include <hip/hip_runtime.h>
#include <hip/hip_bf16.h>

// ---------------------------------------------------------------------------
// CDNA5 (gfx1250) GCN pipeline:
//   - bf16 WMMA (v_wmma_f32_16x16x32_bf16) for the dense x@W GEMMs,
//     fragment-major LDS staging of W, B operands preloaded as ds_load_b128
//   - L2-resident edge gather + global_atomic_add_f32 scatter for message agg
//   - ReLU fused (compile-time) into the next layer's A-operand load
// ---------------------------------------------------------------------------

typedef __attribute__((ext_vector_type(16))) __bf16        v16bf;
typedef __attribute__((ext_vector_type(2)))  __bf16        v2bf;
typedef __attribute__((ext_vector_type(8)))  float         v8f;
typedef __attribute__((ext_vector_type(8)))  unsigned int  v8u;
typedef __attribute__((ext_vector_type(4)))  unsigned int  v4u;

__device__ __forceinline__ unsigned int pack2_bf16(float a, float b) {
#if __has_builtin(__builtin_amdgcn_cvt_pk_bf16_f32)
  // single v_cvt_pk_bf16_f32
  v2bf r = __builtin_amdgcn_cvt_pk_bf16_f32(a, b);
  return __builtin_bit_cast(unsigned int, r);
#else
  // native __bf16 casts: let the backend pick the hw conversion op
  v2bf r = {(__bf16)a, (__bf16)b};
  return __builtin_bit_cast(unsigned int, r);
#endif
}

// ---------------------------------------------------------------------------
// GEMM: h[M,NPAD] = act(x)[M,K] @ W[K,NOUT]   (bf16 WMMA, f32 accum)
// epilogue also writes agg[M,NOUT] = h * dinv[m]^2 + bias   (n < NOUT)
// One wave computes a 16-row strip across all NPAD columns.
// W is staged per 64-K chunk into LDS in fragment-major order:
//   wlds[n*36 + p] = pack_bf16(W[kc+2p][n], W[kc+2p+1][n]),  p in [0,32)
// so a lane's 8 B-fragment dwords (p = ksub*16 + khalf*8 + 0..7) are
// contiguous and 16B-aligned -> two ds_load_b128 per fragment; all NT
// fragments are preloaded before the WMMA chain to hide LDS latency.
// ---------------------------------------------------------------------------
template <int K, int NPAD, int NOUT, bool RELU>
__global__ __launch_bounds__(64) void gcn_gemm(
    const float* __restrict__ x, const float* __restrict__ W,
    const float* __restrict__ bias, const float* __restrict__ dinv,
    float* __restrict__ h, float* __restrict__ agg, int ntiles) {
  constexpr int NT   = NPAD / 16;  // 16-wide N tiles per wave
  constexpr int KCH  = 64;         // K chunk staged in LDS
  constexpr int LROW = 36;         // 32 K-pairs + 4 pad dwords (bank skew)
  __shared__ unsigned int wlds[NPAD * LROW];

  const int lane   = threadIdx.x & 31;
  const int wave   = threadIdx.x >> 5;
  const int tile   = blockIdx.x * 2 + wave;
  const int row0   = tile * 16;
  const int nl     = lane & 15;
  const int khalf  = lane >> 4;
  const bool active = (tile < ntiles);

  v8f zero = {};
  v8f acc[NT];
#pragma unroll
  for (int t = 0; t < NT; ++t) acc[t] = zero;

  const int m = row0 + nl;  // A-fragment row owned by this lane

  for (int kc = 0; kc < K; kc += KCH) {
    __syncthreads();
    // stage W[kc..kc+63][0..NPAD) fragment-major (coalesced W reads over n)
    for (int idx = threadIdx.x; idx < (KCH / 2) * NPAD; idx += 64) {
      const int n = idx % NPAD;
      const int p = idx / NPAD;
      const int k = kc + 2 * p;
      const float w0 = (n < NOUT) ? W[(size_t)k * NOUT + n] : 0.0f;
      const float w1 = (n < NOUT) ? W[(size_t)(k + 1) * NOUT + n] : 0.0f;
      wlds[n * LROW + p] = pack2_bf16(w0, w1);
    }
    __syncthreads();

    if (active) {
#pragma unroll
      for (int ksub = 0; ksub < 2; ++ksub) {
        // A fragment (16x32 bf16): lanes 0-15 rows, contiguous K layout (k = 16*khalf + s)
        v8u aU;
        const float* xr = x + (size_t)m * K + kc + ksub * 32;
#pragma unroll
        for (int j = 0; j < 8; ++j) {
          const int k = khalf * 16 + 2 * j;
          float2 v = *(const float2*)(xr + k);
          if (RELU) {
            v.x = fmaxf(v.x, 0.0f);
            v.y = fmaxf(v.y, 0.0f);
          }
          aU[j] = pack2_bf16(v.x, v.y);
        }
        const v16bf A = __builtin_bit_cast(v16bf, aU);

        // preload ALL B fragments (two ds_load_b128 each), then WMMA chain
        v4u blo[NT], bhi[NT];
#pragma unroll
        for (int t = 0; t < NT; ++t) {
          const unsigned int* bp =
              &wlds[(t * 16 + nl) * LROW + ksub * 16 + khalf * 8];
          blo[t] = *(const v4u*)(bp);
          bhi[t] = *(const v4u*)(bp + 4);
        }
#pragma unroll
        for (int t = 0; t < NT; ++t) {
          const v8u bU =
              __builtin_shufflevector(blo[t], bhi[t], 0, 1, 2, 3, 4, 5, 6, 7);
          const v16bf B = __builtin_bit_cast(v16bf, bU);
          acc[t] = __builtin_amdgcn_wmma_f32_16x16x32_bf16(
              false, A, false, B, (short)0, acc[t], false, false);
        }
      }
    }
  }

  if (!active) return;

  // D layout: lane nl = column, VGPR r -> row r + 8*khalf
#pragma unroll
  for (int t = 0; t < NT; ++t) {
    const int n = t * 16 + nl;
    const float bn = (n < NOUT) ? bias[n] : 0.0f;
#pragma unroll
    for (int r = 0; r < 8; ++r) {
      const int mr = row0 + r + 8 * khalf;
      const float v = acc[t][r];
      h[(size_t)mr * NPAD + n] = v;
      if (n < NOUT) {
        const float di = dinv[mr];
        agg[(size_t)mr * NOUT + n] = v * di * di + bn;
      }
    }
  }
}

// ---------------------------------------------------------------------------
// Edge scatter: agg[dst] += h[src] * dinv[src]*dinv[dst].  One wave per edge,
// scalarized edge state (s_loads), float4 gather per lane (512 B per wave),
// 4 atomics per lane. Working set (h, agg ~51 MB each) is L2-resident.
// ---------------------------------------------------------------------------
__global__ void scatter128(const long long* __restrict__ src,
                           const long long* __restrict__ dst,
                           const float* __restrict__ dinv,
                           const float* __restrict__ h,
                           float* __restrict__ agg, int E) {
  const int lane = threadIdx.x & 31;
  const int e =
      __builtin_amdgcn_readfirstlane(blockIdx.x * 8 + (threadIdx.x >> 5));
  if (e >= E) return;
  const int s = (int)src[e];
  const int d = (int)dst[e];
  const float w = dinv[s] * dinv[d];
  const float4 v = *(const float4*)(h + (size_t)s * 128 + lane * 4);
  float* ap = agg + (size_t)d * 128 + lane * 4;
  atomicAdd(ap + 0, v.x * w);
  atomicAdd(ap + 1, v.y * w);
  atomicAdd(ap + 2, v.z * w);
  atomicAdd(ap + 3, v.w * w);
}

// last layer: h stride 64 (padded), out stride 50
__global__ void scatter_tail(const long long* __restrict__ src,
                             const long long* __restrict__ dst,
                             const float* __restrict__ dinv,
                             const float* __restrict__ h,
                             float* __restrict__ out, int E) {
  const int lane = threadIdx.x & 31;
  const int e =
      __builtin_amdgcn_readfirstlane(blockIdx.x * 8 + (threadIdx.x >> 5));
  if (e >= E) return;
  const int s = (int)src[e];
  const int d = (int)dst[e];
  const float w = dinv[s] * dinv[d];
  for (int c = lane; c < 50; c += 32)
    atomicAdd(out + (size_t)d * 50 + c, h[(size_t)s * 64 + c] * w);
}

// ---------------------------------------------------------------------------
// Degree / dinv / zero helpers
// ---------------------------------------------------------------------------
__global__ void zerof(float* __restrict__ p, int n) {
  const int i = blockIdx.x * blockDim.x + threadIdx.x;
  if (i < n) p[i] = 0.0f;
}
__global__ void deg_accum(const long long* __restrict__ dst,
                          float* __restrict__ deg, int E) {
  const int e = blockIdx.x * blockDim.x + threadIdx.x;
  if (e < E) atomicAdd(&deg[(int)dst[e]], 1.0f);
}
__global__ void make_dinv(float* __restrict__ deg, int n) {
  const int i = blockIdx.x * blockDim.x + threadIdx.x;
  if (i < n) deg[i] = rsqrtf(deg[i] + 1.0f);
}

// ---------------------------------------------------------------------------
extern "C" void kernel_launch(void* const* d_in, const int* in_sizes, int n_in,
                              void* d_out, int out_size, void* d_ws,
                              size_t ws_size, hipStream_t stream) {
  const float* x = (const float*)d_in[0];
  const long long* ei = (const long long*)d_in[1];
  const float* W1 = (const float*)d_in[2];
  const float* b1 = (const float*)d_in[3];
  const float* W2 = (const float*)d_in[4];
  const float* b2 = (const float*)d_in[5];
  const float* W3 = (const float*)d_in[6];
  const float* b3 = (const float*)d_in[7];
  const float* W4 = (const float*)d_in[8];
  const float* b4 = (const float*)d_in[9];
  const float* W5 = (const float*)d_in[10];
  const float* b5 = (const float*)d_in[11];

  const int M = in_sizes[0] / 384;  // 100000
  const int E = in_sizes[1] / 2;    // 1600000
  const long long* src = ei;
  const long long* dst = ei + E;

  // workspace layout (256-byte aligned regions)
  char* base = (char*)d_ws;
  float* dinv = (float*)base;                              // M floats
  float* buf0 = (float*)(base + 524288);                   // M*128 f32
  float* buf1 = (float*)(base + 524288 + 51200000ull);     // M*128 f32
  float* buf2 = (float*)(base + 524288 + 102400000ull);    // M*128 f32
  float* out  = (float*)d_out;                             // M*50 f32

  const int ntiles = M / 16;                // 6250
  const int gemmGrid = (ntiles + 1) / 2;    // 2 waves (tiles) per 64-thr block
  const int scatGrid = (E + 7) / 8;         // 8 waves (edges) per 256-thr block

  // degree + dinv (with self loop)
  zerof<<<(M + 255) / 256, 256, 0, stream>>>(dinv, M);
  deg_accum<<<(E + 255) / 256, 256, 0, stream>>>(dst, dinv, E);
  make_dinv<<<(M + 255) / 256, 256, 0, stream>>>(dinv, M);

  // layer 1: x[.,384] -> buf1   (no input relu)
  gcn_gemm<384, 128, 128, false>
      <<<gemmGrid, 64, 0, stream>>>(x, W1, b1, dinv, buf0, buf1, ntiles);
  scatter128<<<scatGrid, 256, 0, stream>>>(src, dst, dinv, buf0, buf1, E);

  // layer 2: relu(buf1) -> buf2
  gcn_gemm<128, 128, 128, true>
      <<<gemmGrid, 64, 0, stream>>>(buf1, W2, b2, dinv, buf0, buf2, ntiles);
  scatter128<<<scatGrid, 256, 0, stream>>>(src, dst, dinv, buf0, buf2, E);

  // layer 3: relu(buf2) -> buf1
  gcn_gemm<128, 128, 128, true>
      <<<gemmGrid, 64, 0, stream>>>(buf2, W3, b3, dinv, buf0, buf1, ntiles);
  scatter128<<<scatGrid, 256, 0, stream>>>(src, dst, dinv, buf0, buf1, E);

  // layer 4: relu(buf1) -> buf2
  gcn_gemm<128, 128, 128, true>
      <<<gemmGrid, 64, 0, stream>>>(buf1, W4, b4, dinv, buf0, buf2, ntiles);
  scatter128<<<scatGrid, 256, 0, stream>>>(src, dst, dinv, buf0, buf2, E);

  // layer 5: relu(buf2) -> d_out (N=50 padded to 64 for WMMA, no output relu)
  gcn_gemm<128, 64, 50, true>
      <<<gemmGrid, 64, 0, stream>>>(buf2, W5, b5, dinv, buf0, out, ntiles);
  scatter_tail<<<scatGrid, 256, 0, stream>>>(src, dst, dinv, buf0, out, E);
}
